// GNNClassifier_58050777972868
// MI455X (gfx1250) — matrix operand order repaired and checked by reading, not compiled, not run on hardware
//
#include <hip/hip_runtime.h>

// ---------- types ----------
typedef __attribute__((ext_vector_type(16))) __bf16 v16bf;
typedef __attribute__((ext_vector_type(8)))  float  v8f;

union FragU {
  v16bf v;
  unsigned short u[16];
  unsigned int   p[8];
  uint4          q[2];
};

__device__ __forceinline__ unsigned short f2bf(float f) {
  union { float f; unsigned int u; } c; c.f = f;
  unsigned int r = c.u + 0x7FFFu + ((c.u >> 16) & 1u);   // round-to-nearest-even
  return (unsigned short)(r >> 16);
}

// A-matrix (16x32 bf16) per-lane K index, contiguous order used by this code (A and B match):
//   frag elem j (0..15): k = 32*ks + 16*hi + j
__device__ __forceinline__ int kidxA(int j, int hi) {
  return (hi << 4) + j;
}

// Load a 32-byte contiguous bf16 fragment (works for global or LDS pointers;
// B frag elems are k = 32*ks + 16*hi + 0..15, i.e. contiguous in the Wt row).
__device__ __forceinline__ v16bf frag_load(const unsigned short* p) {
  FragU f;
  const uint4* q = (const uint4*)p;   // 16B-aligned by construction
  f.q[0] = q[0];
  f.q[1] = q[1];
  return f.v;
}

// Build A fragment from an fp32 row with the permutation applied (used only on
// the small per-wave LDS mid tile in the head).
__device__ __forceinline__ v16bf frag_a_f32(const float* row, int ks, int hi) {
  FragU f;
#pragma unroll
  for (int j = 0; j < 16; ++j) f.u[j] = f2bf(row[32 * ks + kidxA(j, hi)]);
  return f.v;
}

// ---------- kernels ----------

__global__ void k_zero(float* __restrict__ p, long long n) {
  long long i  = (long long)blockIdx.x * blockDim.x + threadIdx.x;
  long long st = (long long)gridDim.x * blockDim.x;
  for (; i < n; i += st) p[i] = 0.0f;
}

// W [K][N] fp32 -> Wt [Npad][K] bf16 (rows >= N zero-filled)
__global__ void k_w_bf16t(const float* __restrict__ W, unsigned short* __restrict__ Wt,
                          int K, int N, int Npad) {
  int total = Npad * K;
  int st    = gridDim.x * blockDim.x;
  for (int i = blockIdx.x * blockDim.x + threadIdx.x; i < total; i += st) {
    int n = i / K, k = i - n * K;
    float v = (n < N) ? W[(size_t)k * N + n] : 0.0f;
    Wt[i] = f2bf(v);
  }
}

// One wave per edge: gather x[src] (128 f32), scatter-add into sum[dst], count degree.
__global__ __launch_bounds__(256) void k_scatter(const float* __restrict__ x,
                                                 const long long* __restrict__ ei,
                                                 float* __restrict__ sum,
                                                 float* __restrict__ deg, int E) {
  int warp = (int)((blockIdx.x * blockDim.x + threadIdx.x) >> 5);
  int lane = threadIdx.x & 31;
  int nw   = (int)((gridDim.x * blockDim.x) >> 5);
  for (int e = warp; e < E; e += nw) {
    int s = (int)ei[e];
    int d = (int)ei[E + e];
    const float4 xv = *(const float4*)(x + (size_t)s * 128 + lane * 4);
    float* o = sum + (size_t)d * 128 + lane * 4;
    atomicAdd(o + 0, xv.x);
    atomicAdd(o + 1, xv.y);
    atomicAdd(o + 2, xv.z);
    atomicAdd(o + 3, xv.w);
    if (lane == 0) atomicAdd(deg + d, 1.0f);
  }
}

// Convert fp32 activations to bf16 in A-fragment-permuted order:
//   P[row][32*ks + 16*hi + j] = src[row][32*ks + kidxA(j,hi)]
// kidxA maps 4 consecutive j to 4 consecutive k, so each lane moves one float4.
// Lane c handles permuted indices [4c, 4c+4): source k0 = 32*ks + 16*hi + j0.
__global__ __launch_bounds__(256) void k_prep(const float* __restrict__ src,
                                              const float* __restrict__ sum,
                                              const float* __restrict__ deg,
                                              unsigned short* __restrict__ xbf,
                                              unsigned short* __restrict__ meanbf,
                                              int n) {
  int wid  = (int)((blockIdx.x * blockDim.x + threadIdx.x) >> 5);
  int lane = threadIdx.x & 31;
  int nwav = (int)((gridDim.x * blockDim.x) >> 5);
  // permuted index block handled by this lane
  int idx0 = lane * 4;
  int ks   = idx0 >> 5;
  int rem  = idx0 & 31;
  int hi   = rem >> 4;
  int j0   = rem & 15;
  int k0   = 32 * ks + (hi << 4) + j0;
  for (int row = wid; row < n; row += nwav) {
    const float4 xv = *(const float4*)(src + (size_t)row * 128 + k0);
    ushort4 o;
    o.x = f2bf(xv.x); o.y = f2bf(xv.y); o.z = f2bf(xv.z); o.w = f2bf(xv.w);
    *(ushort4*)(xbf + (size_t)row * 128 + idx0) = o;
    if (meanbf) {
      float inv = 1.0f / fmaxf(deg[row], 1.0f);
      const float4 mv = *(const float4*)(sum + (size_t)row * 128 + k0);
      ushort4 mo;
      mo.x = f2bf(mv.x * inv); mo.y = f2bf(mv.y * inv);
      mo.z = f2bf(mv.z * inv); mo.w = f2bf(mv.w * inv);
      *(ushort4*)(meanbf + (size_t)row * 128 + idx0) = mo;
    }
  }
}

// SAGEConv GEMM on pre-permuted bf16 activations, weight-stationary in LDS:
//   out[m] = mean[m] @ Wl + bl + x[m] @ Wr      (one wave per 16-node tile)
__global__ __launch_bounds__(256) void k_sage_gemm(const unsigned short* __restrict__ xbf,
                                                   const unsigned short* __restrict__ meanbf,
                                                   const unsigned short* __restrict__ Wlt,
                                                   const unsigned short* __restrict__ Wrt,
                                                   const float* __restrict__ bl,
                                                   float* __restrict__ out, int ntiles) {
  __shared__ unsigned short wl_s[128 * 128];   // 32 KB
  __shared__ unsigned short wr_s[128 * 128];   // 32 KB
  {
    const uint4* gl = (const uint4*)Wlt;
    const uint4* gr = (const uint4*)Wrt;
    uint4* sl = (uint4*)wl_s;
    uint4* sr = (uint4*)wr_s;
    for (int i = threadIdx.x; i < 2048; i += 256) {
      sl[i] = gl[i];
      sr[i] = gr[i];
    }
  }
  __syncthreads();

  int wid  = (int)((blockIdx.x * blockDim.x + threadIdx.x) >> 5);
  int lane = threadIdx.x & 31;
  int m    = lane & 15, hi = lane >> 4;
  int nwav = (int)((gridDim.x * blockDim.x) >> 5);
  for (int t = wid; t < ntiles; t += nwav) {
    int node0 = t << 4;
    const unsigned short* xrow = xbf    + (size_t)(node0 + m) * 128 + (hi << 4);
    const unsigned short* mrow = meanbf + (size_t)(node0 + m) * 128 + (hi << 4);
    // prefetch next tile's activation rows (global_prefetch_b8)
    int tn = t + nwav;
    if (tn < ntiles) {
      __builtin_prefetch(xbf    + (size_t)((tn << 4) + m) * 128, 0, 1);
      __builtin_prefetch(meanbf + (size_t)((tn << 4) + m) * 128, 0, 1);
    }
    v16bf am[4], ax[4];
#pragma unroll
    for (int ks = 0; ks < 4; ++ks) {
      am[ks] = frag_load(mrow + 32 * ks);
      ax[ks] = frag_load(xrow + 32 * ks);
    }
#pragma unroll 1
    for (int nt = 0; nt < 8; ++nt) {
      int n = (nt << 4) + m;
      const unsigned short* wl = wl_s + n * 128 + (hi << 4);
      const unsigned short* wr = wr_s + n * 128 + (hi << 4);
      v8f acc = {};
#pragma unroll
      for (int ks = 0; ks < 4; ++ks) {
        v16bf bL = frag_load(wl + 32 * ks);
        acc = __builtin_amdgcn_wmma_f32_16x16x32_bf16(false, am[ks], false, bL,
                                                      (short)0, acc, false, false);
        v16bf bR = frag_load(wr + 32 * ks);
        acc = __builtin_amdgcn_wmma_f32_16x16x32_bf16(false, ax[ks], false, bR,
                                                      (short)0, acc, false, false);
      }
      float bias = bl[n];
      float* obase = out + (size_t)(node0 + (hi << 3)) * 128 + n;
#pragma unroll
      for (int j = 0; j < 8; ++j) obase[(size_t)j * 128] = acc[j] + bias;
    }
  }
}

// LayerNorm + ReLU, in place, one wave per node (32 lanes x 4 features).
__global__ __launch_bounds__(256) void k_ln_relu(float* __restrict__ h,
                                                 const float* __restrict__ g,
                                                 const float* __restrict__ b, int n) {
  int wid  = (int)((blockIdx.x * blockDim.x + threadIdx.x) >> 5);
  int lane = threadIdx.x & 31;
  int nwav = (int)((gridDim.x * blockDim.x) >> 5);
  for (int row = wid; row < n; row += nwav) {
    float4 v = *(float4*)(h + (size_t)row * 128 + lane * 4);
    float s  = v.x + v.y + v.z + v.w;
    float s2 = v.x * v.x + v.y * v.y + v.z * v.z + v.w * v.w;
#pragma unroll
    for (int o = 16; o > 0; o >>= 1) {
      s  += __shfl_xor(s,  o, 32);
      s2 += __shfl_xor(s2, o, 32);
    }
    float mu  = s * (1.0f / 128.0f);
    float var = s2 * (1.0f / 128.0f) - mu * mu;
    float inv = rsqrtf(var + 1e-5f);
    float4 gg = *(const float4*)(g + lane * 4);
    float4 bb = *(const float4*)(b + lane * 4);
    v.x = fmaxf((v.x - mu) * inv * gg.x + bb.x, 0.0f);
    v.y = fmaxf((v.y - mu) * inv * gg.y + bb.y, 0.0f);
    v.z = fmaxf((v.z - mu) * inv * gg.z + bb.z, 0.0f);
    v.w = fmaxf((v.w - mu) * inv * gg.w + bb.w, 0.0f);
    *(float4*)(h + (size_t)row * 128 + lane * 4) = v;
  }
}

// Head: relu(hbf @ Wc1 + bc1) @ Wc2 + bc2 ; one wave per 16-node tile.
// GEMM1 A comes from the permuted bf16 buffer; mid tile staged through per-wave
// LDS to re-layout WMMA C -> A fragments for GEMM2.
__global__ __launch_bounds__(256) void k_head(const unsigned short* __restrict__ hbf,
                                              const unsigned short* __restrict__ Wc1t,
                                              const float* __restrict__ bc1,
                                              const unsigned short* __restrict__ Wc2t,
                                              const float* __restrict__ bc2,
                                              float* __restrict__ out, int ntiles) {
  __shared__ float mid[8][16 * 64];   // 32 KB: per-wave 16x64 fp32 tile
  int wib  = threadIdx.x >> 5;
  int wid  = (int)((blockIdx.x * blockDim.x + threadIdx.x) >> 5);
  int lane = threadIdx.x & 31;
  int m    = lane & 15, hi = lane >> 4;
  int nwav = (int)((gridDim.x * blockDim.x) >> 5);
  float* mymid = mid[wib];
  for (int t = wid; t < ntiles; t += nwav) {
    int node0 = t << 4;
    const unsigned short* hrow = hbf + (size_t)(node0 + m) * 128 + (hi << 4);
    v16bf a[4];
#pragma unroll
    for (int ks = 0; ks < 4; ++ks) a[ks] = frag_load(hrow + 32 * ks);
#pragma unroll 1
    for (int nt = 0; nt < 4; ++nt) {          // 64 mid cols
      int n = (nt << 4) + m;
      const unsigned short* w = Wc1t + (size_t)n * 128 + (hi << 4);
      v8f acc = {};
#pragma unroll
      for (int ks = 0; ks < 4; ++ks) {
        v16bf bf = frag_load(w + 32 * ks);
        acc = __builtin_amdgcn_wmma_f32_16x16x32_bf16(false, a[ks], false, bf,
                                                      (short)0, acc, false, false);
      }
      float bias = bc1[n];
#pragma unroll
      for (int j = 0; j < 8; ++j)
        mymid[(j + (hi << 3)) * 64 + n] = fmaxf(acc[j] + bias, 0.0f);
    }
    // second GEMM: 16x64 (bf16) @ 64x48 -> 16x48, keep first 40 cols
    const float* mrow = mymid + m * 64;
    v16bf a2[2];
#pragma unroll
    for (int ks = 0; ks < 2; ++ks) a2[ks] = frag_a_f32(mrow, ks, hi);
#pragma unroll 1
    for (int nt = 0; nt < 3; ++nt) {
      int n = (nt << 4) + m;
      const unsigned short* w = Wc2t + (size_t)n * 64 + (hi << 4);
      v8f acc = {};
#pragma unroll
      for (int ks = 0; ks < 2; ++ks) {
        v16bf bf = frag_load(w + 32 * ks);
        acc = __builtin_amdgcn_wmma_f32_16x16x32_bf16(false, a2[ks], false, bf,
                                                      (short)0, acc, false, false);
      }
      if (n < 40) {
        float bias = bc2[n];
#pragma unroll
        for (int j = 0; j < 8; ++j)
          out[(size_t)(node0 + j + (hi << 3)) * 40 + n] = acc[j] + bias;
      }
    }
  }
}

// ---------- host ----------
extern "C" void kernel_launch(void* const* d_in, const int* in_sizes, int n_in,
                              void* d_out, int out_size, void* d_ws, size_t ws_size,
                              hipStream_t stream) {
  const float*     x   = (const float*)d_in[0];
  const long long* ei  = (const long long*)d_in[1];
  const float*     Wl1 = (const float*)d_in[2];
  const float*     bl1 = (const float*)d_in[3];
  const float*     Wr1 = (const float*)d_in[4];
  const float*     g1  = (const float*)d_in[5];
  const float*     be1 = (const float*)d_in[6];
  const float*     Wl2 = (const float*)d_in[7];
  const float*     bl2 = (const float*)d_in[8];
  const float*     Wr2 = (const float*)d_in[9];
  const float*     g2  = (const float*)d_in[10];
  const float*     be2 = (const float*)d_in[11];
  const float*     Wc1 = (const float*)d_in[12];
  const float*     bc1 = (const float*)d_in[13];
  const float*     Wc2 = (const float*)d_in[14];
  const float*     bc2 = (const float*)d_in[15];
  float* out = (float*)d_out;

  const int Nn = in_sizes[0] / 128;    // 100000
  const int Ee = in_sizes[1] / 2;      // 625000
  const int ntiles = Nn / 16;          // 6250 (N divisible by 16)

  // workspace carve-out
  char* ws = (char*)d_ws;
  size_t off = 0;
  auto carve = [&](size_t bytes) -> void* {
    void* p = ws + off;
    off = (off + bytes + 255) & ~(size_t)255;
    return p;
  };
  float* sum = (float*)carve((size_t)Nn * 129 * sizeof(float)); // sum[N*128] + deg[N]
  float* deg = sum + (size_t)Nn * 128;
  float*          hbuf   = (float*)carve((size_t)Nn * 128 * sizeof(float));
  unsigned short* xbf    = (unsigned short*)carve((size_t)Nn * 128 * 2);
  unsigned short* meanbf = (unsigned short*)carve((size_t)Nn * 128 * 2);
  unsigned short* wl1t = (unsigned short*)carve(128 * 128 * 2);
  unsigned short* wr1t = (unsigned short*)carve(128 * 128 * 2);
  unsigned short* wl2t = (unsigned short*)carve(128 * 128 * 2);
  unsigned short* wr2t = (unsigned short*)carve(128 * 128 * 2);
  unsigned short* wc1t = (unsigned short*)carve(64 * 128 * 2);
  unsigned short* wc2t = (unsigned short*)carve(48 * 64 * 2);

  // 1) weight conversion (tiny)
  k_w_bf16t<<<64, 256, 0, stream>>>(Wl1, wl1t, 128, 128, 128);
  k_w_bf16t<<<64, 256, 0, stream>>>(Wr1, wr1t, 128, 128, 128);
  k_w_bf16t<<<64, 256, 0, stream>>>(Wl2, wl2t, 128, 128, 128);
  k_w_bf16t<<<64, 256, 0, stream>>>(Wr2, wr2t, 128, 128, 128);
  k_w_bf16t<<<64, 256, 0, stream>>>(Wc1, wc1t, 128, 64, 64);
  k_w_bf16t<<<32, 256, 0, stream>>>(Wc2, wc2t, 64, 40, 48);

  const int gemmBlocks = 512;                    // 4096 waves, ~1.5 tiles each
  const int rowBlocks  = (Nn * 32 + 255) / 256;

  // 2) layer 1
  k_zero<<<8192, 256, 0, stream>>>(sum, (long long)Nn * 129);
  k_scatter<<<8192, 256, 0, stream>>>(x, ei, sum, deg, Ee);
  k_prep<<<rowBlocks, 256, 0, stream>>>(x, sum, deg, xbf, meanbf, Nn);
  k_sage_gemm<<<gemmBlocks, 256, 0, stream>>>(xbf, meanbf, wl1t, wr1t, bl1, hbuf, ntiles);
  k_ln_relu<<<rowBlocks, 256, 0, stream>>>(hbuf, g1, be1, Nn);

  // 3) layer 2
  k_zero<<<8192, 256, 0, stream>>>(sum, (long long)Nn * 129);
  k_scatter<<<8192, 256, 0, stream>>>(hbuf, ei, sum, deg, Ee);
  k_prep<<<rowBlocks, 256, 0, stream>>>(hbuf, sum, deg, xbf, meanbf, Nn);
  k_sage_gemm<<<gemmBlocks, 256, 0, stream>>>(xbf, meanbf, wl2t, wr2t, bl2, hbuf, ntiles);
  k_ln_relu<<<rowBlocks, 256, 0, stream>>>(hbuf, g2, be2, Nn);

  // 4) classifier head (prep h -> permuted bf16, reuse xbf)
  k_prep<<<rowBlocks, 256, 0, stream>>>(hbuf, nullptr, nullptr, xbf, nullptr, Nn);
  k_head<<<gemmBlocks, 256, 0, stream>>>(xbf, wc1t, bc1, wc2t, bc2, out, ntiles);
}
